// MultiHeadAttention_8899172237483
// MI455X (gfx1250) — matrix operand order repaired and checked by reading, not compiled, not run on hardware
//
#include <hip/hip_runtime.h>
#include <hip/hip_bf16.h>

typedef __attribute__((ext_vector_type(16))) _Float16 v16h;
typedef __attribute__((ext_vector_type(8)))  float    v8f;
typedef unsigned int u32;

#define HID   512
#define NHEAD 8
#define HD    64
#define SEQ   2048
#define BATCH 4

union H2U   { u32 u; _Float16 h[2]; };
union FragA { v16h v; u32 u[8]; };

// CDNA5 wave32 WMMA f16 fragment index helpers (ISA 7.12.2)
// A 16x32: lane L -> M = L%16 ; VGPR v holds K = (v<4?0:16) + (L/16)*8 + (v&3)*2, +1
__device__ __forceinline__ int a_kbase(int v, int hl) {
    return (v < 4 ? 0 : 16) + hl * 8 + (v & 3) * 2;
}
// B 32x16: lane L -> N = L%16 ; VGPR v holds K = (v<4?0:16) + (L/16)*8 + (v&3)*2, +1
__device__ __forceinline__ int b_kbase(int v, int hl) {
    return (v < 4 ? 0 : 16) + hl * 8 + (v & 3) * 2;
}

// CDNA5 async global->LDS copy, 16B per lane, tracked by ASYNCcnt.
// lds_off = wave-relative LDS byte address (low 32 bits of generic pointer).
__device__ __forceinline__ void async_copy_b128(u32 lds_off, const void* gaddr) {
    asm volatile("global_load_async_to_lds_b128 %0, %1, off"
                 :: "v"(lds_off), "v"(gaddr) : "memory");
}
__device__ __forceinline__ void wait_async_le4() {
    asm volatile("s_wait_asynccnt 0x4" ::: "memory");
}
__device__ __forceinline__ void wait_async_le0() {
    asm volatile("s_wait_asynccnt 0x0" ::: "memory");
}

// ---------------------------------------------------------------------------
// Tiled GEMM: Y[M=8192, 512] = X[8192,512] @ W[512,512] + bias
// WG = 256 threads (8 waves). 64x64 tile, each wave computes 16x32.
// A_F16: X is f16 (ctx) else f32. OUT_F32: write f32 to Yf else f16 to Yh.
// ---------------------------------------------------------------------------
template<bool A_F16, bool OUT_F32>
__launch_bounds__(256)
__global__ void gemm512_kernel(const void* __restrict__ Ap,
                               const float* __restrict__ W,
                               const float* __restrict__ bias,
                               _Float16* __restrict__ Yh,
                               float* __restrict__ Yf)
{
    __shared__ _Float16 As[64][40];   // 64x32 + pad (row = 80B, 16B aligned)
    __shared__ _Float16 Bs[32][72];   // 32x64 + pad (row = 144B, 16B aligned)

    const int tid  = threadIdx.x;
    const int wv   = tid >> 5;
    const int lane = tid & 31;
    const int hl   = lane >> 4;
    const int l16  = lane & 15;

    const int n0 = blockIdx.x * 64;
    const int m0 = blockIdx.y * 64;
    const int mW = (wv & 3) * 16;
    const int nW = (wv >> 2) * 32;

    v8f acc0 = {};
    v8f acc1 = {};

    const int arow = tid >> 2;        // 0..63
    const int akg  = (tid & 3) * 8;   // 0..24
    const int bk   = tid >> 3;        // 0..31
    const int bng  = (tid & 7) * 8;   // 0..56

    for (int k0 = 0; k0 < HID; k0 += 32) {
        __syncthreads();
        // --- stage A tile (convert to f16) ---
        {
            _Float16 t[8];
            if (A_F16) {
                const uint4* p = (const uint4*)((const _Float16*)Ap +
                                   (size_t)(m0 + arow) * HID + k0 + akg);
                uint4 q = *p;
                H2U c;
                c.u = q.x; t[0]=c.h[0]; t[1]=c.h[1];
                c.u = q.y; t[2]=c.h[0]; t[3]=c.h[1];
                c.u = q.z; t[4]=c.h[0]; t[5]=c.h[1];
                c.u = q.w; t[6]=c.h[0]; t[7]=c.h[1];
            } else {
                const float4* p = (const float4*)((const float*)Ap +
                                   (size_t)(m0 + arow) * HID + k0 + akg);
                float4 x = p[0], y = p[1];
                t[0]=(_Float16)x.x; t[1]=(_Float16)x.y; t[2]=(_Float16)x.z; t[3]=(_Float16)x.w;
                t[4]=(_Float16)y.x; t[5]=(_Float16)y.y; t[6]=(_Float16)y.z; t[7]=(_Float16)y.w;
            }
            u32* d = (u32*)&As[arow][akg];
            H2U c;
            #pragma unroll
            for (int i = 0; i < 4; ++i) { c.h[0]=t[2*i]; c.h[1]=t[2*i+1]; d[i]=c.u; }
        }
        // --- stage B tile (W is f32 row-major [k][n]) ---
        {
            const float4* p = (const float4*)(W + (size_t)(k0 + bk) * HID + n0 + bng);
            float4 x = p[0], y = p[1];
            _Float16 t[8] = {(_Float16)x.x,(_Float16)x.y,(_Float16)x.z,(_Float16)x.w,
                             (_Float16)y.x,(_Float16)y.y,(_Float16)y.z,(_Float16)y.w};
            u32* d = (u32*)&Bs[bk][bng];
            H2U c;
            #pragma unroll
            for (int i = 0; i < 4; ++i) { c.h[0]=t[2*i]; c.h[1]=t[2*i+1]; d[i]=c.u; }
        }
        __syncthreads();

        // --- build fragments, issue WMMA ---
        FragA a;
        #pragma unroll
        for (int v = 0; v < 8; ++v)
            a.u[v] = *(const u32*)&As[mW + l16][a_kbase(v, hl)];

        FragA b0, b1;
        #pragma unroll
        for (int v = 0; v < 8; ++v) {
            const int kb = b_kbase(v, hl);
            H2U c;
            c.h[0] = Bs[kb][nW + l16];      c.h[1] = Bs[kb+1][nW + l16];      b0.u[v] = c.u;
            c.h[0] = Bs[kb][nW + 16 + l16]; c.h[1] = Bs[kb+1][nW + 16 + l16]; b1.u[v] = c.u;
        }
        acc0 = __builtin_amdgcn_wmma_f32_16x16x32_f16(false, a.v, false, b0.v,
                                                      (short)0, acc0, false, false);
        acc1 = __builtin_amdgcn_wmma_f32_16x16x32_f16(false, a.v, false, b1.v,
                                                      (short)0, acc1, false, false);
    }

    // --- epilogue: C layout M = vgpr + 8*(lane/16), N = lane%16 ---
    #pragma unroll
    for (int r = 0; r < 8; ++r) {
        const int row = m0 + mW + hl * 8 + r;
        const int col = n0 + nW + l16;
        const float v0 = acc0[r] + bias[col];
        const float v1 = acc1[r] + bias[col + 16];
        if (OUT_F32) {
            Yf[(size_t)row * HID + col]      = v0;
            Yf[(size_t)row * HID + col + 16] = v1;
        } else {
            Yh[(size_t)row * HID + col]      = (_Float16)v0;
            Yh[(size_t)row * HID + col + 16] = (_Float16)v1;
        }
    }
}

// ---------------------------------------------------------------------------
// Flash attention: WG = 256 threads (8 waves) per (batch, head, 128-query tile)
// Each wave owns 16 query rows. K/V streamed in 64-key chunks, double-buffered
// in LDS via CDNA5 async global->LDS copies (ASYNCcnt pipeline).
// ---------------------------------------------------------------------------
__launch_bounds__(256)
__global__ void attn_kernel(const _Float16* __restrict__ Qf,
                            const _Float16* __restrict__ Kf,
                            const _Float16* __restrict__ Vf,
                            const int* __restrict__ mask,
                            _Float16* __restrict__ Ctx)
{
    __shared__ _Float16 Kt[2][64][72];   // [buf][key][d] + pad (row = 144B)
    __shared__ _Float16 Vt[2][64][72];
    __shared__ _Float16 Pl[8][16][68];   // per-wave P scratch (C->A relayout)
    __shared__ float    maskAdd[64];

    const int tid  = threadIdx.x;
    const int wv   = tid >> 5;           // 0..7
    const int lane = tid & 31;
    const int hl   = lane >> 4;
    const int l16  = lane & 15;

    const int bb = blockIdx.z;
    const int hh = blockIdx.y;
    const int q0 = blockIdx.x * 128;
    const int c0 = hh * HD;
    const size_t rowBase = (size_t)bb * SEQ + q0;

    // Q fragments (A layout), d-halves 0..31 and 32..63 — loaded once
    FragA qa0, qa1;
    {
        const _Float16* qrow = Qf + (rowBase + wv * 16 + l16) * HID + c0;
        #pragma unroll
        for (int v = 0; v < 8; ++v) {
            const int kb = a_kbase(v, hl);
            qa0.u[v] = *(const u32*)(qrow + kb);
            qa1.u[v] = *(const u32*)(qrow + kb + 32);
        }
    }

    float mrow[8], lrow[8];
    #pragma unroll
    for (int r = 0; r < 8; ++r) { mrow[r] = -3.0e38f; lrow[r] = 0.0f; }
    v8f o0 = {}, o1 = {}, o2 = {}, o3 = {};

    const float scale = 0.125f;          // 1/sqrt(64)

    // staging map: 256 threads x 32B = one 64x64 f16 chunk per matrix
    const int lr  = tid >> 2;            // 0..63 : chunk row
    const int ldg = (tid & 3) * 16;      // 0,16,32,48 : d offset

    // issue one chunk's async copies: 4 ops/wave (2 K + 2 V), 16B/lane each
    auto issue_chunk = [&](int j0, int buf) {
        const size_t srow = ((size_t)bb * SEQ + j0 + lr) * HID + c0 + ldg;
        const u32 koff = (u32)(size_t)&Kt[buf][lr][ldg];
        const u32 voff = (u32)(size_t)&Vt[buf][lr][ldg];
        async_copy_b128(koff,      Kf + srow);
        async_copy_b128(koff + 16, Kf + srow + 8);
        async_copy_b128(voff,      Vf + srow);
        async_copy_b128(voff + 16, Vf + srow + 8);
    };

    issue_chunk(0, 0);                   // prologue: chunk 0 in flight

    for (int j0 = 0, it = 0; j0 < SEQ; j0 += 64, ++it) {
        const int  buf     = it & 1;
        const bool hasNext = (j0 + 64 < SEQ);

        // prefetch next chunk into the other buffer (prev barrier made it free)
        if (hasNext) issue_chunk(j0 + 64, buf ^ 1);
        if (tid < 64)
            maskAdd[tid] = (mask[bb * SEQ + j0 + tid] == 0) ? -1.0e9f : 0.0f;
        // wait for current chunk only: <=4 leaves next chunk in flight
        if (hasNext) wait_async_le4(); else wait_async_le0();
        __syncthreads();

        const _Float16 (* __restrict__ Ktc)[72] = Kt[buf];
        const _Float16 (* __restrict__ Vtc)[72] = Vt[buf];

        // --- scores: S = Q @ K^T, 4 key tiles of 16 ---
        v8f s0 = {}, s1 = {}, s2 = {}, s3 = {};
        auto scoreTile = [&](v8f& sAcc, int t) {
            FragA b0, b1;
            const int n = t * 16 + l16;              // local key = B's N
            #pragma unroll
            for (int v = 0; v < 8; ++v) {
                const int kb = b_kbase(v, hl);       // K index = head dim d
                b0.u[v] = *(const u32*)&Ktc[n][kb];
                b1.u[v] = *(const u32*)&Ktc[n][kb + 32];
            }
            sAcc = __builtin_amdgcn_wmma_f32_16x16x32_f16(false, qa0.v, false, b0.v,
                                                          (short)0, sAcc, false, false);
            sAcc = __builtin_amdgcn_wmma_f32_16x16x32_f16(false, qa1.v, false, b1.v,
                                                          (short)0, sAcc, false, false);
        };
        scoreTile(s0, 0); scoreTile(s1, 1); scoreTile(s2, 2); scoreTile(s3, 3);

        // --- scale + mask + online softmax (rows live in lane-halves) ---
        float corr[8];
        #pragma unroll
        for (int r = 0; r < 8; ++r) {
            float v0 = s0[r] * scale + maskAdd[ 0 + l16];
            float v1 = s1[r] * scale + maskAdd[16 + l16];
            float v2 = s2[r] * scale + maskAdd[32 + l16];
            float v3 = s3[r] * scale + maskAdd[48 + l16];
            float mx = fmaxf(fmaxf(v0, v1), fmaxf(v2, v3));
            #pragma unroll
            for (int d = 1; d < 16; d <<= 1)       // xor<16 stays within half
                mx = fmaxf(mx, __shfl_xor(mx, d, 32));
            const float mnew = fmaxf(mrow[r], mx);
            const float cr   = __expf(mrow[r] - mnew);
            v0 = __expf(v0 - mnew); v1 = __expf(v1 - mnew);
            v2 = __expf(v2 - mnew); v3 = __expf(v3 - mnew);
            float rs = v0 + v1 + v2 + v3;
            #pragma unroll
            for (int d = 1; d < 16; d <<= 1)
                rs += __shfl_xor(rs, d, 32);
            lrow[r] = lrow[r] * cr + rs;
            mrow[r] = mnew;
            corr[r] = cr;
            // spill P (C layout) into per-wave LDS for A-layout reload
            const int M = hl * 8 + r;
            Pl[wv][M][ 0 + l16] = (_Float16)v0;
            Pl[wv][M][16 + l16] = (_Float16)v1;
            Pl[wv][M][32 + l16] = (_Float16)v2;
            Pl[wv][M][48 + l16] = (_Float16)v3;
        }
        #pragma unroll
        for (int r = 0; r < 8; ++r) {
            o0[r] *= corr[r]; o1[r] *= corr[r];
            o2[r] *= corr[r]; o3[r] *= corr[r];
        }

        // --- O += P @ V, 4 d tiles ---
        FragA pa0, pa1;
        #pragma unroll
        for (int v = 0; v < 8; ++v) {
            const int kb = a_kbase(v, hl);
            pa0.u[v] = *(const u32*)&Pl[wv][l16][kb];
            pa1.u[v] = *(const u32*)&Pl[wv][l16][kb + 32];
        }
        auto pvTile = [&](v8f& oAcc, int t) {
            FragA b0, b1;
            const int n = t * 16 + l16;              // local d = B's N
            #pragma unroll
            for (int v = 0; v < 8; ++v) {
                const int kb = b_kbase(v, hl);       // K index = local key
                H2U c;
                c.h[0] = Vtc[kb][n];      c.h[1] = Vtc[kb + 1][n];  b0.u[v] = c.u;
                c.h[0] = Vtc[kb + 32][n]; c.h[1] = Vtc[kb + 33][n]; b1.u[v] = c.u;
            }
            oAcc = __builtin_amdgcn_wmma_f32_16x16x32_f16(false, pa0.v, false, b0.v,
                                                          (short)0, oAcc, false, false);
            oAcc = __builtin_amdgcn_wmma_f32_16x16x32_f16(false, pa1.v, false, b1.v,
                                                          (short)0, oAcc, false, false);
        };
        pvTile(o0, 0); pvTile(o1, 1); pvTile(o2, 2); pvTile(o3, 3);

        __syncthreads();   // all waves done with buf before it is re-filled
    }

    // --- normalize + store ctx (merged-head f16 layout [b*S+s][h*64+d]) ---
    #pragma unroll
    for (int r = 0; r < 8; ++r) {
        const float inv = 1.0f / lrow[r];
        const size_t row = rowBase + wv * 16 + hl * 8 + r;
        _Float16* crow = Ctx + row * HID + c0;
        crow[ 0 + l16] = (_Float16)(o0[r] * inv);
        crow[16 + l16] = (_Float16)(o1[r] * inv);
        crow[32 + l16] = (_Float16)(o2[r] * inv);
        crow[48 + l16] = (_Float16)(o3[r] * inv);
    }
}

// ---------------------------------------------------------------------------
extern "C" void kernel_launch(void* const* d_in, const int* in_sizes, int n_in,
                              void* d_out, int out_size, void* d_ws, size_t ws_size,
                              hipStream_t stream)
{
    const float* values = (const float*)d_in[0];
    const float* keys   = (const float*)d_in[1];
    const float* query  = (const float*)d_in[2];
    const int*   mask   = (const int*)d_in[3];
    const float* Wq = (const float*)d_in[4];   const float* bq = (const float*)d_in[5];
    const float* Wk = (const float*)d_in[6];   const float* bk = (const float*)d_in[7];
    const float* Wv = (const float*)d_in[8];   const float* bv = (const float*)d_in[9];
    const float* Wo = (const float*)d_in[10];  const float* bo = (const float*)d_in[11];

    const size_t M    = (size_t)BATCH * SEQ;   // 8192
    const size_t NELT = M * HID;               // 4,194,304 elements per tensor

    _Float16* Qh = (_Float16*)d_ws;
    _Float16* Kh = Qh + NELT;
    _Float16* Vh = Kh + NELT;
    // ctx may alias Q: each (b,h,qtile) block reads only its own rows/columns
    // of Q (at kernel start) and writes the identical region of ctx (at end).
    _Float16* Ctx = (ws_size >= 4 * NELT * sizeof(_Float16)) ? (Vh + NELT) : Qh;

    const dim3 gB(256), gG(HID / 64, M / 64);
    gemm512_kernel<false, false><<<gG, gB, 0, stream>>>(query,  Wq, bq, Qh, nullptr);
    gemm512_kernel<false, false><<<gG, gB, 0, stream>>>(keys,   Wk, bk, Kh, nullptr);
    gemm512_kernel<false, false><<<gG, gB, 0, stream>>>(values, Wv, bv, Vh, nullptr);

    const dim3 aB(256), aG(SEQ / 128, NHEAD, BATCH);
    attn_kernel<<<aG, aB, 0, stream>>>(Qh, Kh, Vh, mask, Ctx);

    gemm512_kernel<true, true><<<gG, gB, 0, stream>>>(Ctx, Wo, bo, nullptr, (float*)d_out);
}
